// FlowFunc_23871428231761
// MI455X (gfx1250) — matrix operand order repaired and checked by reading, not compiled, not run on hardware
//
#include <hip/hip_runtime.h>

// CDNA5 / gfx1250 fused local-correlation soft-argmax flow, two-stage:
//
//  1) pack_f16_pad_kernel: transpose (n,c,s,h,w) f32 -> channel-last,
//     ZERO-PADDED-HALO (b, y+3, x+3, c) f16 in d_ws (WP=64, HP=70; 9.2 MB,
//     L2-resident). Every WMMA fragment becomes a contiguous, always
//     in-bounds 16/32-byte run -> pure global_load_b128 feeds with zero
//     boundary masking in the hot loop.
//
//  2) flow_local_corr_wmma: one wave32 per (batch, row, 16-px x-tile).
//     corr(16 px x 49 taps) = 7(dy) x 2(N-tile) x 2(K-step)
//     v_wmma_f32_16x16x32_f16 shifted-Gram GEMMs; diagonal band extracted
//     with a branchless dummy-slot LDS scatter (addresses hoisted out of the
//     dy loop); lanes 0-15 do the masked softmax + soft-argmax.

typedef __attribute__((ext_vector_type(16))) _Float16 v16h;
typedef __attribute__((ext_vector_type(8)))  _Float16 v8h;
typedef __attribute__((ext_vector_type(8)))  float    v8f;

namespace {
constexpr int N_  = 2;               // batch n
constexpr int C_  = 64;              // channels
constexpr int S_  = 4;               // sequence
constexpr int H_  = 64;
constexpr int W_  = 44;
constexpr int R_  = 3;               // window radius
constexpr int KT_ = 2 * R_ + 1;      // 7
constexpr int KK_ = KT_ * KT_;       // 49 taps
constexpr int TILE = 16;
constexpr int XT_  = (W_ + TILE - 1) / TILE; // 3 x-tiles
constexpr int B_   = N_ * S_;                // 8 fused batches
constexpr int CSTR = S_ * H_ * W_;           // f32 channel stride
constexpr int WP   = 64;                     // padded width  (x+R in [0,63])
constexpr int HP   = H_ + 2 * R_;            // padded height (70)
constexpr int PIXP = B_ * HP * WP;           // 35840 padded pixels
constexpr int CSTRIDE = KK_ + 7;             // padded corr row (56)
}

// ---------------------------------------------------------------------------
// Stage 1: (n,c,s,h,w) f32 -> zero-padded channel-last (b, y+R, x+R, c) f16.
// One thread per PADDED pixel; halo threads store zeros.  Reads: lanes along
// x -> coalesced per channel.  Writes: 128 contiguous bytes per lane.
// ---------------------------------------------------------------------------
__global__ __launch_bounds__(256)
void pack_f16_pad_kernel(const float* __restrict__ fin, _Float16* __restrict__ fout)
{
  const int p = blockIdx.x * blockDim.x + threadIdx.x;    // (b*HP + yp)*WP + xp
  if (p >= PIXP) return;
  const int xp  = p % WP;
  const int rem = p / WP;
  const int yp  = rem % HP;
  const int b   = rem / HP;
  const int x   = xp - R_;
  const int y   = yp - R_;

  v8h* ov = (v8h*)(fout + (long)p * C_);         // 16B-aligned (128B/pixel)
  if (x >= 0 && x < W_ && y >= 0 && y < H_) {
    const int nidx = b / S_;
    const int sidx = b % S_;
    const long inbase = ((long)(nidx * C_) * S_ + sidx) * (long)(H_ * W_)
                      + (long)y * W_ + x;
#pragma unroll
    for (int g = 0; g < 8; ++g) {
      v8h pack;
#pragma unroll
      for (int j = 0; j < 8; ++j)
        pack[j] = (_Float16)fin[inbase + (long)(g * 8 + j) * CSTR];
      ov[g] = pack;
    }
  } else {
    const v8h z = {};
#pragma unroll
    for (int g = 0; g < 8; ++g) ov[g] = z;       // halo -> zeros
  }
}

// ---------------------------------------------------------------------------
// Stage 2: fused local correlation + softmax + soft-argmax.
// ---------------------------------------------------------------------------
__global__ __launch_bounds__(32)
void flow_local_corr_wmma(const _Float16* __restrict__ g0,   // padded f16 f0
                          const _Float16* __restrict__ g1,   // padded f16 f1
                          float* __restrict__ out)
{
  const int lane = threadIdx.x;      // wave32 lane
  const int nloc = lane & 15;        // column within 16-wide fragment
  const int hiH  = lane >> 4;        // 0: lanes 0-15, 1: lanes 16-31
  const int x0   = blockIdx.x * TILE;
  const int y    = blockIdx.y;
  const int b    = blockIdx.z;       // nidx*S + sidx
  const int nidx = b / S_;
  const int sidx = b % S_;

  // rows 0..15 = real corr rows; row 16 = dummy spill slot.
  __shared__ float corrF[(TILE + 1) * CSTRIDE];

  // ---------------- A fragments: f0 tile (16 px x 64 ch), reused 7x -------
  // A k order used here (same as B): lanes 0-15 hold K 0..15, lanes 16-31
  // K 16..31: one contiguous 16-half run -> two b128 loads.
  // Halo padding makes every load in-bounds and zero outside the image.
  const _Float16* pA =
      g0 + (((long)(b * HP + (y + R_)) * WP) + (x0 + nloc + R_)) * C_;
  v16h aF[2];
#pragma unroll
  for (int kt = 0; kt < 2; ++kt) {
    const int cb = kt * 32 + hiH * 16;
    const v8h lo = *(const v8h*)(pA + cb);        // K 0..7   (+16 hi lanes)
    const v8h hi = *(const v8h*)(pA + cb + 8);    // K 8..15  (+16 hi lanes)
    aF[kt] = __builtin_shufflevector(lo, hi, 0, 1, 2, 3, 4, 5, 6, 7,
                                             8, 9, 10, 11, 12, 13, 14, 15);
  }

  // ---- hoisted branchless band-scatter pattern (dy-invariant) -------------
  // For a given (lane, i) at most ONE of the two N-tiles is in the ±R band
  // (dx1 = dx0 + 16).  Out-of-band slots target the dummy row; ky offsets
  // (0..42) keep the dummy stores inside the dummy row (<= 42 < 56).
  int  sbase[8];
  bool sel0[8];
#pragma unroll
  for (int i = 0; i < 8; ++i) {
    const int  m   = hiH * 8 + i;
    const int  dx0 = nloc - m - R_;               // j0 = 0 tile
    const int  dx1 = dx0 + 16;                    // j0 = 16 tile
    const bool in0 = (dx0 >= -R_) && (dx0 <= R_);
    const bool in1 = (dx1 >= -R_) && (dx1 <= R_);
    sel0[i]  = in0;
    sbase[i] = (in0 || in1) ? (m * CSTRIDE + (in0 ? dx0 : dx1) + R_)
                            : (TILE * CSTRIDE);
  }

  // ---------------- per-dy shifted Gram tiles via WMMA --------------------
#pragma unroll 1
  for (int dy = -R_; dy <= R_; ++dy) {
    // padded column for j0=0 is x0 + nloc (x0 - R + nloc, then +R halo shift)
    const _Float16* pB0 =
        g1 + (((long)(b * HP + (y + R_ + dy)) * WP) + (x0 + nloc)) * C_;
    const _Float16* pB1 = pB0 + 16 * C_;          // j0 = 16 tile

    v8f acc0 = {};   // columns j = 0..15  of the 22-wide union window
    v8f acc1 = {};   // columns j = 16..21 (padded to 31)
#pragma unroll
    for (int kt = 0; kt < 2; ++kt) {
      // B k order used here (same as A): lanes 0-15 K=0..15, lanes 16-31 K=16..31:
      // one contiguous 16-half run -> two b128 loads, no masking needed.
      const int cb = kt * 32 + hiH * 16;
      const v8h a0 = *(const v8h*)(pB0 + cb);
      const v8h a1 = *(const v8h*)(pB0 + cb + 8);
      const v8h b0 = *(const v8h*)(pB1 + cb);
      const v8h b1 = *(const v8h*)(pB1 + cb + 8);
      const v16h bLo = __builtin_shufflevector(a0, a1, 0, 1, 2, 3, 4, 5, 6, 7,
                                                       8, 9, 10, 11, 12, 13, 14, 15);
      const v16h bHi = __builtin_shufflevector(b0, b1, 0, 1, 2, 3, 4, 5, 6, 7,
                                                       8, 9, 10, 11, 12, 13, 14, 15);
      acc0 = __builtin_amdgcn_wmma_f32_16x16x32_f16(false, aF[kt], false, bLo,
                                                    (short)0, acc0, false, false);
      acc1 = __builtin_amdgcn_wmma_f32_16x16x32_f16(false, aF[kt], false, bHi,
                                                    (short)0, acc1, false, false);
    }

    // branchless diagonal-band scatter: corr[m,dx] = G[m, m+dx+R] / sqrt(C)
    const int ky = (dy + R_) * KT_;
#pragma unroll
    for (int i = 0; i < 8; ++i) {
      const float val = (sel0[i] ? acc0[i] : acc1[i]) * 0.125f;
      corrF[sbase[i] + ky] = val;                 // unconditional ds_store
    }
  }

  __syncthreads();

  // ---------------- masked softmax + soft-argmax (lanes 0-15) -------------
  if (lane < TILE) {
    const int x = x0 + lane;
    if (x < W_) {
      float v[KK_];
      float mx = -3.4e38f;
#pragma unroll
      for (int t = 0; t < KK_; ++t) {
        const int dy = t / KT_ - R_;
        const int dx = t % KT_ - R_;
        const bool ok = (x + dx >= 0) && (x + dx < W_) &&
                        (y + dy >= 0) && (y + dy < H_);
        float cv = corrF[lane * CSTRIDE + t];     // unconditional ds_load
        cv = ok ? cv : -3.4e38f;                  // -inf at OOB taps
        v[t] = cv;
        mx = fmaxf(mx, cv);
      }
      float sum = 0.0f, ex = 0.0f, ey = 0.0f;
#pragma unroll
      for (int t = 0; t < KK_; ++t) {
        const int dy = t / KT_ - R_;
        const int dx = t % KT_ - R_;
        const float e = __expf(v[t] - mx);
        sum += e;
        ex  += e * (float)dx;
        ey  += e * (float)dy;
      }
      const float inv = 1.0f / sum;
      // output layout (n, 2, s, h, w); flow = E[sample] - base = E[(dx,dy)]
      const long o0 = ((((long)nidx * 2 + 0) * S_ + sidx) * H_ + y) * (long)W_ + x;
      const long o1 = ((((long)nidx * 2 + 1) * S_ + sidx) * H_ + y) * (long)W_ + x;
      out[o0] = ex * inv;
      out[o1] = ey * inv;
    }
  }
}

extern "C" void kernel_launch(void* const* d_in, const int* in_sizes, int n_in,
                              void* d_out, int out_size, void* d_ws, size_t ws_size,
                              hipStream_t stream) {
  (void)in_sizes; (void)n_in; (void)out_size; (void)ws_size;
  const float* f0 = (const float*)d_in[0];
  const float* f1 = (const float*)d_in[1];
  float* out = (float*)d_out;

  _Float16* ws0 = (_Float16*)d_ws;                // padded (b,yp,xp,c) f16, f0
  _Float16* ws1 = ws0 + (size_t)PIXP * C_;        // padded (b,yp,xp,c) f16, f1

  // Stage 1: pack both feature maps to padded channel-last f16 (9.2 MB).
  const int packBlocks = (PIXP + 255) / 256;
  pack_f16_pad_kernel<<<packBlocks, 256, 0, stream>>>(f0, ws0);
  pack_f16_pad_kernel<<<packBlocks, 256, 0, stream>>>(f1, ws1);

  // Stage 2: fused correlation + softmax + flow.
  dim3 grid(XT_, H_, B_);        // (3, 64, 8)
  dim3 block(32);                // one wave32 per tile
  flow_local_corr_wmma<<<grid, block, 0, stream>>>(ws0, ws1, out);
}
